// projection_net_28089086116226
// MI455X (gfx1250) — compile-verified, matrix operand order checked by reading, not run on hardware
//
#include <hip/hip_runtime.h>

// ---------------- types ----------------
typedef __bf16 v16bf __attribute__((ext_vector_type(16)));
typedef float  v8f   __attribute__((ext_vector_type(8)));
typedef int    v4i   __attribute__((ext_vector_type(4)));   // matches async-LDS builtin param

struct U8x4 { uint4 lo; uint4 hi; };

static __device__ inline v16bf make_frag(uint4 lo, uint4 hi) {
  U8x4 t{lo, hi};
  return __builtin_bit_cast(v16bf, t);
}

static __device__ inline unsigned short f2bf(float f) {
  unsigned u = __float_as_uint(f);
  unsigned r = (u + 0x7FFFu + ((u >> 16) & 1u)) >> 16;
  return (unsigned short)r;
}

// Async memory->LDS path (CDNA5): GLOBAL_LOAD_ASYNC_TO_LDS_B128 + ASYNCcnt.
#if defined(__AMDGCN__) && __has_builtin(__builtin_amdgcn_global_load_async_to_lds_b128)
#define USE_ASYNC_LDS 1
#else
#define USE_ASYNC_LDS 0
#endif

static __device__ inline void wait_async0() {
#if defined(__AMDGCN__) && __has_builtin(__builtin_amdgcn_s_wait_asynccnt)
  __builtin_amdgcn_s_wait_asynccnt(0);
#else
  asm volatile("s_wait_asynccnt 0x0" ::: "memory");
#endif
}

// ---------------- sizes ----------------
#define BATCH 4096
#define HDIM  512
#define NBASE 64
#define BM    128
#define BN    128
#define KC    32
#define RSPL  2
#define RPB   (NBASE / RSPL)   // 32 bases per block
#define BSTRIDE 40             // padded halfword stride for B tile rows (80B, 16B aligned)
#define ASTRIDE 33             // padded float stride for att tile

// ============================================================
// Kernel 1: rel_base f32 [r][i][o]  ->  bf16 [r][o][i] (K-contiguous)
// ============================================================
__global__ __launch_bounds__(256) void prep_rbt(const float* __restrict__ rb,
                                                unsigned short* __restrict__ rbt) {
  __shared__ float tile[32][33];
  const int r  = blockIdx.z;
  const int i0 = blockIdx.y * 32;
  const int o0 = blockIdx.x * 32;
  const int tx = threadIdx.x & 31;
  const int ty = threadIdx.x >> 5;   // 0..7
#pragma unroll
  for (int k = 0; k < 4; k++) {
    int row = ty + k * 8;
    tile[row][tx] = rb[((size_t)r * HDIM + (i0 + row)) * HDIM + o0 + tx];
  }
  __syncthreads();
#pragma unroll
  for (int k = 0; k < 4; k++) {
    int row = ty + k * 8;
    rbt[((size_t)r * HDIM + (o0 + row)) * HDIM + i0 + tx] = f2bf(tile[tx][row]);
  }
}

// ============================================================
// Kernel 2: emb f32 -> bf16, att gather
// ============================================================
__global__ __launch_bounds__(256) void prep_misc(const float* __restrict__ emb,
                                                 const int* __restrict__ ids,
                                                 const float* __restrict__ rel_att,
                                                 unsigned short* __restrict__ embw,
                                                 float* __restrict__ attw) {
  int g = blockIdx.x * 256 + threadIdx.x;   // 0 .. 2M-1
  embw[g] = f2bf(emb[g]);
  if (g < BATCH * NBASE) {
    int b = g >> 6, rr = g & 63;
    attw[g] = rel_att[(size_t)ids[b] * NBASE + rr];
  }
}

// ============================================================
// Kernel 3: fused GEMM  out[b,o] = sum_r att[b,r] * (emb[b,:] @ RB[r,:,:])[o]
//   A = emb (bf16, scaled per-r by att in bf16), B = RBT (bf16, [r][o][i])
//   split-K over bases -> 2 deterministic partial buffers
//   B tiles staged via GLOBAL_LOAD_ASYNC_TO_LDS_B128 (ASYNCcnt) when available
// ============================================================
__global__ __launch_bounds__(256) void gemm_kernel(const unsigned short* __restrict__ RBT,
                                                   const unsigned short* __restrict__ embw,
                                                   const float* __restrict__ attw,
                                                   float* __restrict__ outp) {
  __shared__ __align__(16) unsigned short BsT[2][BN * BSTRIDE];  // 20 KB
  __shared__ float attS[BM * ASTRIDE];                           // ~16.5 KB

  const int tid  = threadIdx.x;
  const int lane = tid & 31;
  const int wave = tid >> 5;
  const int wm   = wave & 3;   // wave row  (0..3) -> 32 rows each
  const int wn   = wave >> 2;  // wave col  (0..1) -> 64 cols each
  const int m0   = blockIdx.x * BM;
  const int n0   = blockIdx.y * BN;
  const int r0   = blockIdx.z * RPB;

  // preload att tile: attS[row][rr], padded stride
  for (int idx = tid; idx < BM * RPB; idx += 256) {
    int row = idx >> 5;          // /RPB
    int rr  = idx & (RPB - 1);
    attS[row * ASTRIDE + rr] = attw[(size_t)(m0 + row) * NBASE + r0 + rr];
  }

  const int lm   = lane & 15;
  const int lh   = lane >> 4;    // 0/1
  const int koff = lh * 8;       // K offset per documented 16-bit A/B layout

  v8f acc[2][4];
  v8f zacc = {};
#pragma unroll
  for (int mi = 0; mi < 2; mi++)
#pragma unroll
    for (int ni = 0; ni < 4; ni++) acc[mi][ni] = zacc;

  // staging mapping: thread covers 32 contiguous bytes of one B-tile row
  const int so  = tid >> 1;          // row n (0..127)
  const int suc = (tid & 1) * 2;     // unit-of-8 offset {0,2}

  __syncthreads();

  for (int i0 = 0; i0 < HDIM; i0 += KC) {
    // ---- raw A fragments for this K-chunk (reused across all r) ----
    uint4 araw[2][2];
#pragma unroll
    for (int mi = 0; mi < 2; mi++) {
      int row = m0 + wm * 32 + mi * 16 + lm;
      const uint4* ap = (const uint4*)(embw + (size_t)row * HDIM + i0 + koff);
      araw[mi][0] = ap[0];   // k = koff .. koff+7
      araw[mi][1] = ap[2];   // k = 16+koff .. 16+koff+7
    }
    // ---- stage first B tile (r = r0) ----
    {
      const unsigned short* gsrc =
          RBT + ((size_t)r0 * HDIM + (n0 + so)) * HDIM + i0 + suc * 8;
      unsigned short* ldst = &BsT[0][so * BSTRIDE + suc * 8];
#if USE_ASYNC_LDS
      __builtin_amdgcn_global_load_async_to_lds_b128((v4i*)gsrc, (v4i*)ldst, 0, 0);
      __builtin_amdgcn_global_load_async_to_lds_b128((v4i*)(gsrc + 8), (v4i*)(ldst + 8), 0, 0);
      wait_async0();
#else
      const uint4* gp = (const uint4*)gsrc;
      uint4 q0 = gp[0], q1 = gp[1];
      uint4* lp = (uint4*)ldst;
      lp[0] = q0; lp[1] = q1;
#endif
    }
    __syncthreads();

    for (int rr = 0; rr < RPB; rr++) {
      const int cur = rr & 1;
      // prefetch next B tile into the other buffer
      if (rr + 1 < RPB) {
        const unsigned short* gsrc =
            RBT + ((size_t)(r0 + rr + 1) * HDIM + (n0 + so)) * HDIM + i0 + suc * 8;
        unsigned short* ldst = &BsT[cur ^ 1][so * BSTRIDE + suc * 8];
#if USE_ASYNC_LDS
        __builtin_amdgcn_global_load_async_to_lds_b128((v4i*)gsrc, (v4i*)ldst, 0, 0);
        __builtin_amdgcn_global_load_async_to_lds_b128((v4i*)(gsrc + 8), (v4i*)(ldst + 8), 0, 0);
#else
        const uint4* gp = (const uint4*)gsrc;
        uint4 q0 = gp[0], q1 = gp[1];
        uint4* lp = (uint4*)ldst;
        lp[0] = q0; lp[1] = q1;
#endif
      }
      // ---- B fragments from LDS (b128, conflict-free padded stride) ----
      v16bf bf[4];
#pragma unroll
      for (int ni = 0; ni < 4; ni++) {
        int col = wn * 64 + ni * 16 + lm;
        uint4 b0 = *(const uint4*)(&BsT[cur][col * BSTRIDE + koff]);
        uint4 b1 = *(const uint4*)(&BsT[cur][col * BSTRIDE + 16 + koff]);
        bf[ni] = make_frag(b0, b1);
      }
      // ---- scale A by att[row, r] (bf16 splat mul) and accumulate ----
#pragma unroll
      for (int mi = 0; mi < 2; mi++) {
        float attv = attS[(wm * 32 + mi * 16 + lm) * ASTRIDE + rr];
        __bf16 ab = (__bf16)attv;
        v16bf as = make_frag(araw[mi][0], araw[mi][1]) * ab;  // v_pk_mul_bf16
#pragma unroll
        for (int ni = 0; ni < 4; ni++) {
          acc[mi][ni] = __builtin_amdgcn_wmma_f32_16x16x32_bf16(
              false, as, false, bf[ni], (short)0, acc[mi][ni], false, false);
        }
      }
#if USE_ASYNC_LDS
      wait_async0();   // issued async tile complete before the joint barrier
#endif
      __syncthreads();
    }
  }

  // ---- epilogue: write split-K partial (deterministic, no atomics) ----
  float* op = outp + (size_t)blockIdx.z * ((size_t)BATCH * HDIM);
#pragma unroll
  for (int mi = 0; mi < 2; mi++) {
    int rbase = m0 + wm * 32 + mi * 16 + lh * 8;   // C layout: M = v (+8 for hi lanes)
#pragma unroll
    for (int ni = 0; ni < 4; ni++) {
      int col = n0 + wn * 64 + ni * 16 + lm;       // C layout: N = lane&15
#pragma unroll
      for (int v = 0; v < 8; v++) {
        op[(size_t)(rbase + v) * HDIM + col] = acc[mi][ni][v];
      }
    }
  }
}

// ============================================================
// Kernel 4: y = LayerNorm(part0 + part1 + att @ rel_bias)
// ============================================================
__global__ __launch_bounds__(256) void ln_kernel(const float* __restrict__ gemw,
                                                 const float* __restrict__ attw,
                                                 const float* __restrict__ rel_bias,
                                                 float* __restrict__ out) {
  __shared__ float attL[NBASE];
  __shared__ float red[256];
  __shared__ float red2[256];
  __shared__ float stats[2];
  const int b = blockIdx.x;
  const int tid = threadIdx.x;
  if (tid < NBASE) attL[tid] = attw[(size_t)b * NBASE + tid];
  __syncthreads();

  float v[2];
#pragma unroll
  for (int j = 0; j < 2; j++) {
    int c = tid + j * 256;
    float x = gemw[(size_t)b * HDIM + c] +
              gemw[(size_t)BATCH * HDIM + (size_t)b * HDIM + c];
    float bias = 0.f;
#pragma unroll 8
    for (int r = 0; r < NBASE; r++) bias += attL[r] * rel_bias[(size_t)r * HDIM + c];
    v[j] = x + bias;
  }
  red[tid]  = v[0] + v[1];
  red2[tid] = v[0] * v[0] + v[1] * v[1];
  __syncthreads();
  for (int st = 128; st > 0; st >>= 1) {
    if (tid < st) { red[tid] += red[tid + st]; red2[tid] += red2[tid + st]; }
    __syncthreads();
  }
  if (tid == 0) {
    float mu  = red[0] * (1.0f / HDIM);
    float var = red2[0] * (1.0f / HDIM) - mu * mu;
    stats[0] = mu;
    stats[1] = rsqrtf(var + 1e-5f);
  }
  __syncthreads();
  float mu = stats[0], inv = stats[1];
#pragma unroll
  for (int j = 0; j < 2; j++) {
    int c = tid + j * 256;
    out[(size_t)b * HDIM + c] = (v[j] - mu) * inv;
  }
}

// ============================================================
// launch
// ============================================================
extern "C" void kernel_launch(void* const* d_in, const int* in_sizes, int n_in,
                              void* d_out, int out_size, void* d_ws, size_t ws_size,
                              hipStream_t stream) {
  (void)in_sizes; (void)n_in; (void)out_size; (void)ws_size;
  const float* emb      = (const float*)d_in[0];
  const int*   ids      = (const int*)d_in[1];
  const float* rel_att  = (const float*)d_in[2];
  const float* rel_base = (const float*)d_in[3];
  const float* rel_bias = (const float*)d_in[4];
  float* out = (float*)d_out;

  char* ws = (char*)d_ws;
  unsigned short* rbt  = (unsigned short*)(ws);                       // 32 MB  bf16 [r][o][i]
  unsigned short* embw = (unsigned short*)(ws + (size_t)33554432);    //  4 MB  bf16 [b][i]
  float*          attw = (float*)(ws + (size_t)37748736);             //  1 MB  f32  [b][r]
  float*          gemw = (float*)(ws + (size_t)38797312);             // 16 MB  f32  2x[b][o]

  prep_rbt   <<<dim3(16, 16, 64), 256, 0, stream>>>(rel_base, rbt);
  prep_misc  <<<dim3(8192),       256, 0, stream>>>(emb, ids, rel_att, embw, attw);
  gemm_kernel<<<dim3(32, 4, 2),   256, 0, stream>>>(rbt, embw, attw, gemw);
  ln_kernel  <<<dim3(4096),       256, 0, stream>>>(gemw, attw, rel_bias, out);
}
